// SlideAttention_1726576854818
// MI455X (gfx1250) — matrix operand order repaired and checked by reading, not compiled, not run on hardware
//
#include <hip/hip_runtime.h>

// ---------------------------------------------------------------------------
// SlideAttention for MI455X (gfx1250, wave32, WMMA bf16 16x16x32 f32-accum,
// async global->LDS staging on the attention stage)
// ---------------------------------------------------------------------------

typedef __attribute__((ext_vector_type(16))) __bf16 v16bf;
typedef __attribute__((ext_vector_type(8)))  float  v8f;

#define B_     16
#define DIM_   256
#define NH_    8
#define CH_    8          // DIM/DR/NH
#define NQKV_  192        // 3*DIM/DR
#define HS_    56
#define HW_    3136
#define QK_SCALE 0.17677669529663687f   // (DIM/NH)^-0.5 = 1/sqrt(32)

// fp32 -> bf16, round-to-nearest-even (pure integer ops, always compiles)
__device__ __forceinline__ unsigned short f2bf(float f) {
  unsigned u = __float_as_uint(f);
  u += 0x7FFFu + ((u >> 16) & 1u);
  return (unsigned short)(u >> 16);
}

union FragA { uint4 q[2]; v16bf v; };   // 32 bytes = 16 bf16 = 8 VGPRs

// CDNA5 async global->LDS copy (ASYNCcnt-tracked, bypasses VGPRs).
// The clang builtin is prototyped with int* operands (addrspace(1) source,
// addrspace(3) destination). Guarded: falls back to a plain load/store path
// if the builtin is absent (host pass, or older toolchains).
#if __has_builtin(__builtin_amdgcn_global_load_async_to_lds_b32) && \
    __has_builtin(__builtin_amdgcn_s_wait_asynccnt)
#define HAVE_ASYNC_LDS 1
#define AS1_ __attribute__((address_space(1)))
#define AS3_ __attribute__((address_space(3)))
__device__ __forceinline__ void async_g2l_b32(const float* g, float* l) {
  __builtin_amdgcn_global_load_async_to_lds_b32(
      (AS1_ int*)(uintptr_t)g, (AS3_ int*)(uintptr_t)l, 0, 0);
}
#endif

// ---------------------------------------------------------------------------
// Kernel 1: qkv = x^T @ qkv_w^T + qkv_b, fused NCHW transpose.
//   A: M = spatial (per image), K = 256 input channels. x is [B][C][HW], so
//      A^T (k-major) is contiguous in m -> coalesced loads, bf16 LDS restage.
//   B: qkv_w [n][k] row-major -> per-lane K{0..7,16..23} / K{8..15,24..31} by lane half (ISA B layout).
//   Output f stored plane-major [B][192][HW] for the attention kernel.
// ---------------------------------------------------------------------------
__global__ __launch_bounds__(128)
void qkv_gemm_kernel(const float* __restrict__ x,
                     const float* __restrict__ qkv_w,
                     const float* __restrict__ qkv_b,
                     float* __restrict__ f)
{
  __shared__ unsigned short Bs[NQKV_][DIM_];   // 96 KB bf16 weights [n][k]
  __shared__ unsigned short As[64][32];        // 4 KB bf16 A tile  [m][k]

  const int tid = threadIdx.x;
  const int b   = blockIdx.y;
  const int m0  = blockIdx.x * 64;

  // Stage full weight matrix as bf16 into LDS (reused by the whole K loop).
  for (int i = tid; i < NQKV_ * DIM_; i += 128)
    (&Bs[0][0])[i] = f2bf(qkv_w[i]);

  const int wave  = tid >> 5;
  const int lane  = tid & 31;
  const int l16   = lane & 15;
  const int lhalf = lane >> 4;

  v8f acc[12];
#pragma unroll
  for (int nt = 0; nt < 12; ++nt) {
    float bias = qkv_b[nt * 16 + l16];
#pragma unroll
    for (int r = 0; r < 8; ++r) acc[nt][r] = bias;
  }

  const float* xb = x + (size_t)b * DIM_ * HW_;

  for (int k0 = 0; k0 < DIM_; k0 += 32) {
    __syncthreads();
    // A tile: 64 rows (spatial) x 32 K (channels); contiguous in m per k.
    {
      const int m  = tid & 63;
      const int kb = tid >> 6;   // 0..1
#pragma unroll
      for (int kk = 0; kk < 32; kk += 2) {
        const int k = kk + kb;
        As[m][k] = f2bf(xb[(size_t)(k0 + k) * HW_ + (m0 + m)]);
      }
    }
    // CDNA5 prefetch of next A tile (lowers to global_prefetch_b8).
    if (k0 + 32 < DIM_)
      __builtin_prefetch(&xb[(size_t)(k0 + 32 + (tid >> 6)) * HW_ + m0 + (tid & 63)], 0, 1);
    __syncthreads();

    // Preload ALL fragments into distinct registers first, so the 26
    // ds_load_b128 can be in flight together and the 12 WMMAs issue
    // back-to-back behind a single s_wait_dscnt.
    FragA af;
    {
      // A fragment (16-bit A 16x32 layout): lane<16 -> K{0..7,16..23},
      // lane>=16 -> K{8..15,24..31}; row m = wave*16 + (lane&15).
      const int m   = wave * 16 + l16;
      const int klo = lhalf * 8;
      af.q[0] = *(const uint4*)&As[m][klo];
      af.q[1] = *(const uint4*)&As[m][klo + 16];
    }
    FragA bfr[12];
#pragma unroll
    for (int nt = 0; nt < 12; ++nt) {
      // B fragment (32x16): lane = col n (mod 16), same K split as A.
      const int n  = nt * 16 + l16;
      const int ks = k0 + lhalf * 8;
      bfr[nt].q[0] = *(const uint4*)&Bs[n][ks];
      bfr[nt].q[1] = *(const uint4*)&Bs[n][ks + 16];
    }
#pragma unroll
    for (int nt = 0; nt < 12; ++nt)
      acc[nt] = __builtin_amdgcn_wmma_f32_16x16x32_bf16(
          false, af.v, false, bfr[nt].v, (short)0, acc[nt], false, false);
  }

  // D layout: lane<16 -> N=lane, M=vgpr(0..7); lane>=16 -> N=lane-16, M=8+vgpr.
  // Each lane's 8 results are 8 consecutive m at one n -> two aligned b128 stores.
  float* fb = f + (size_t)b * NQKV_ * HW_;
  const int mrow = m0 + wave * 16 + lhalf * 8;
#pragma unroll
  for (int nt = 0; nt < 12; ++nt) {
    const int n = nt * 16 + l16;
    float* p = fb + (size_t)n * HW_ + mrow;
    float4 s0 = {acc[nt][0], acc[nt][1], acc[nt][2], acc[nt][3]};
    float4 s1 = {acc[nt][4], acc[nt][5], acc[nt][6], acc[nt][7]};
    *(float4*)(p)     = s0;
    *(float4*)(p + 4) = s1;
  }
}

// ---------------------------------------------------------------------------
// Kernel 2: fused unfold + depthwise-conv branch + rpb + softmax(9) + v-sum.
// One block per (b, head, 14x14 tile); k/v haloed 16x16 patches staged into
// LDS via CDNA5 async global->LDS loads (zero-pad via disjoint LDS stores).
// K=9 makes this a VALU job, not a WMMA job.
// ---------------------------------------------------------------------------
__global__ __launch_bounds__(256)
void attn_kernel(const float* __restrict__ f,
                 const float* __restrict__ dc_b,
                 const float* __restrict__ dc1_w,
                 const float* __restrict__ dc1_b,
                 const float* __restrict__ rpb,
                 float* __restrict__ y)
{
  __shared__ float kbuf[CH_][16][16];   // 8 KB
  __shared__ float vbuf[CH_][16][16];   // 8 KB
  __shared__ float w9[72][9];           // dc1_w, 2.6 KB
  __shared__ float wb[72];              // dc_b + dc1_b folded
  __shared__ float rp[9];               // rpb for this head

  const int tid = threadIdx.x;
  const int bh  = blockIdx.z;
  const int b   = bh >> 3, h = bh & 7;
  const int y0  = blockIdx.y * 14, x0 = blockIdx.x * 14;

  for (int i = tid; i < 72 * 9; i += 256) w9[i / 9][i % 9] = dc1_w[i];
  for (int i = tid; i < 72;     i += 256) wb[i] = dc_b[i] + dc1_b[i];
  if (tid < 9) rp[tid] = rpb[h * 9 + tid];

  const float* qpl = f + ((size_t)b * NQKV_ + h * 24) * HW_;
  const float* kpl = qpl + (size_t)CH_ * HW_;
  const float* vpl = qpl + (size_t)(2 * CH_) * HW_;

  // Stage haloed 16x16 patches for all 8 channels (zero-pad OOB).
  for (int e = tid; e < CH_ * 256; e += 256) {
    const int c = e >> 8, pos = e & 255;
    const int py = pos >> 4, px = pos & 15;
    const int gy = y0 + py - 1, gx = x0 + px - 1;
    const bool in = (gy >= 0) & (gy < HS_) & (gx >= 0) & (gx < HS_);
    const size_t off = (size_t)c * HW_ + gy * HS_ + gx;
#ifdef HAVE_ASYNC_LDS
    if (in) {
      // direct memory->LDS, no VGPR round-trip; per-lane EXEC predication
      async_g2l_b32(kpl + off, &kbuf[c][py][px]);
      async_g2l_b32(vpl + off, &vbuf[c][py][px]);
    } else {
      kbuf[c][py][px] = 0.f;   // disjoint addresses from the async writes
      vbuf[c][py][px] = 0.f;
    }
#else
    kbuf[c][py][px] = in ? kpl[off] : 0.f;
    vbuf[c][py][px] = in ? vpl[off] : 0.f;
#endif
  }
#ifdef HAVE_ASYNC_LDS
  __builtin_amdgcn_s_wait_asynccnt(0);
#endif
  __syncthreads();

  if (tid >= 196) return;
  const int ly = tid / 14, lx = tid % 14;
  const int hw = (y0 + ly) * HS_ + (x0 + lx);

  float q[CH_], qsum = 0.f;
#pragma unroll
  for (int c = 0; c < CH_; ++c) {
    q[c] = qpl[(size_t)c * HW_ + hw] * QK_SCALE;
    qsum += q[c];
  }

  // rpb broadcasts over channels: (q . rpb_j) = rpb_j * sum_c q_c
  float logit[9];
#pragma unroll
  for (int j = 0; j < 9; ++j) logit[j] = qsum * rp[j];

#pragma unroll
  for (int c = 0; c < CH_; ++c) {
    float p[9];
#pragma unroll
    for (int dy = 0; dy < 3; ++dy)
#pragma unroll
      for (int dx = 0; dx < 3; ++dx)
        p[dy * 3 + dx] = kbuf[c][ly + dy][lx + dx];
    const float qc = q[c];
#pragma unroll
    for (int j = 0; j < 9; ++j) {
      const float* w = w9[c * 9 + j];
      float conv = 0.f;
#pragma unroll
      for (int t = 0; t < 9; ++t) conv = fmaf(w[t], p[t], conv);
      logit[j] += qc * (p[j] + wb[c * 9 + j] + conv);
    }
  }

  // softmax over the 9 taps
  float mx = logit[0];
#pragma unroll
  for (int j = 1; j < 9; ++j) mx = fmaxf(mx, logit[j]);
  float a[9], s = 0.f;
#pragma unroll
  for (int j = 0; j < 9; ++j) { a[j] = __expf(logit[j] - mx); s += a[j]; }
  const float inv = 1.f / s;
#pragma unroll
  for (int j = 0; j < 9; ++j) a[j] *= inv;

  // weighted v-sum; output pixel-major [B][HW][64] for the proj GEMM.
  float* yo = y + ((size_t)(b * HW_ + hw) * 64) + h * CH_;
#pragma unroll
  for (int c = 0; c < CH_; ++c) {
    float p[9];
#pragma unroll
    for (int dy = 0; dy < 3; ++dy)
#pragma unroll
      for (int dx = 0; dx < 3; ++dx)
        p[dy * 3 + dx] = vbuf[c][ly + dy][lx + dx];
    float o = 0.f;
#pragma unroll
    for (int j = 0; j < 9; ++j) {
      const float* w = w9[c * 9 + j];
      float conv = 0.f;
#pragma unroll
      for (int t = 0; t < 9; ++t) conv = fmaf(w[t], p[t], conv);
      o = fmaf(a[j], p[j] + wb[c * 9 + j] + conv, o);
    }
    yo[c] = o;
  }
}

// ---------------------------------------------------------------------------
// Kernel 3: out = y @ proj_w^T + proj_b, output NCHW-plane-major.
// M=50176, N=256, K=64 (2 WMMA K-steps). Block: 8 waves = 4 Mx2 N split.
// ---------------------------------------------------------------------------
__global__ __launch_bounds__(256)
void proj_gemm_kernel(const float* __restrict__ y,
                      const float* __restrict__ proj_w,
                      const float* __restrict__ proj_b,
                      float* __restrict__ out)
{
  __shared__ unsigned short Bs[DIM_][64];   // 32 KB bf16 proj_w [n][k]
  __shared__ unsigned short As[64][64];     // 8 KB bf16 A tile  [m][k]

  const int tid = threadIdx.x;
  const int b   = blockIdx.y;
  const int m0  = blockIdx.x * 64;

  for (int i = tid; i < DIM_ * 64; i += 256)
    (&Bs[0][0])[i] = f2bf(proj_w[i]);

  // A tile: y is [M][64] row-major -> fully coalesced.
  {
    const int m  = tid >> 2;
    const int ks = (tid & 3) * 16;
    const float* src = y + ((size_t)(b * HW_ + m0 + m) * 64) + ks;
#pragma unroll
    for (int i = 0; i < 16; ++i) As[m][ks + i] = f2bf(src[i]);
  }
  __syncthreads();

  const int wave  = tid >> 5;
  const int lane  = tid & 31;
  const int l16   = lane & 15;
  const int lhalf = lane >> 4;
  const int mw    = (wave & 3) * 16;
  const int n0    = (wave >> 2) * 128;

  v8f acc[8];
#pragma unroll
  for (int nt = 0; nt < 8; ++nt) {
    float bias = proj_b[n0 + nt * 16 + l16];
#pragma unroll
    for (int r = 0; r < 8; ++r) acc[nt][r] = bias;
  }

#pragma unroll
  for (int k0 = 0; k0 < 64; k0 += 32) {
    FragA af;
    const int m   = mw + l16;
    const int klo = k0 + lhalf * 8;
    af.q[0] = *(const uint4*)&As[m][klo];
    af.q[1] = *(const uint4*)&As[m][klo + 16];
    FragA bfr[8];
#pragma unroll
    for (int nt = 0; nt < 8; ++nt) {
      const int n  = n0 + nt * 16 + l16;
      const int ks = k0 + lhalf * 8;
      bfr[nt].q[0] = *(const uint4*)&Bs[n][ks];
      bfr[nt].q[1] = *(const uint4*)&Bs[n][ks + 16];
    }
#pragma unroll
    for (int nt = 0; nt < 8; ++nt)
      acc[nt] = __builtin_amdgcn_wmma_f32_16x16x32_bf16(
          false, af.v, false, bfr[nt].v, (short)0, acc[nt], false, false);
  }

  float* ob = out + (size_t)b * DIM_ * HW_;
  const int mrow = m0 + mw + lhalf * 8;
#pragma unroll
  for (int nt = 0; nt < 8; ++nt) {
    const int n = n0 + nt * 16 + l16;
    float* p = ob + (size_t)n * HW_ + mrow;
    float4 s0 = {acc[nt][0], acc[nt][1], acc[nt][2], acc[nt][3]};
    float4 s1 = {acc[nt][4], acc[nt][5], acc[nt][6], acc[nt][7]};
    *(float4*)(p)     = s0;
    *(float4*)(p + 4) = s1;
  }
}

// ---------------------------------------------------------------------------
extern "C" void kernel_launch(void* const* d_in, const int* in_sizes, int n_in,
                              void* d_out, int out_size, void* d_ws, size_t ws_size,
                              hipStream_t stream) {
  const float* x      = (const float*)d_in[0];
  const float* qkv_w  = (const float*)d_in[1];
  const float* qkv_b  = (const float*)d_in[2];
  const float* dc_b   = (const float*)d_in[3];
  const float* dc1_w  = (const float*)d_in[4];
  const float* dc1_b  = (const float*)d_in[5];
  const float* rpb    = (const float*)d_in[6];
  const float* proj_w = (const float*)d_in[7];
  const float* proj_b = (const float*)d_in[8];
  float* outp = (float*)d_out;

  // scratch: f [16][192][3136] fp32 (38.5 MB), y [16][3136][64] fp32 (12.8 MB)
  float* f  = (float*)d_ws;
  float* yb = f + (size_t)B_ * NQKV_ * HW_;

  qkv_gemm_kernel<<<dim3(49, B_), 128, 0, stream>>>(x, qkv_w, qkv_b, f);
  attn_kernel<<<dim3(4, 4, B_ * NH_), 256, 0, stream>>>(f, dc_b, dc1_w, dc1_b, rpb, yb);
  proj_gemm_kernel<<<dim3(49, B_), 256, 0, stream>>>(yb, proj_w, proj_b, outp);
}
